// TemporalAttention_51230369906785
// MI455X (gfx1250) — matrix operand order repaired and checked by reading, not compiled, not run on hardware
//
#include <hip/hip_runtime.h>
#include <hip/hip_bf16.h>
#include <cstdint>

#define B_SZ   16
#define T_SZ   24
#define N_SZ   325
#define D_SZ   512
#define DQKV   1536
#define K_HDS  8
#define D_HEAD 64
#define TPAD   32
#define M_ROWS (B_SZ * T_SZ * N_SZ)   /* 124800 */
#define BHN    (B_SZ * K_HDS * N_SZ)  /* 41600  */

typedef __attribute__((ext_vector_type(8)))  __bf16 v8bf;
typedef __attribute__((ext_vector_type(16))) __bf16 v16bf;
typedef __attribute__((ext_vector_type(8)))  float  v8f;

static __device__ __forceinline__ __bf16 f2bf(float f) {
  unsigned u = __builtin_bit_cast(unsigned, f);
  u += 0x7FFFu + ((u >> 16) & 1u);               // round-to-nearest-even
  unsigned short h = (unsigned short)(u >> 16);
  return __builtin_bit_cast(__bf16, h);
}

static __device__ __forceinline__ v16bf cat8(v8bf a, v8bf b) {
  return __builtin_shufflevector(a, b, 0,1,2,3,4,5,6,7,8,9,10,11,12,13,14,15);
}

// Per-lane async DMA: 16 bytes global -> LDS, tracked by ASYNCcnt (no VGPR
// landing, no loadcnt serialization).  VDST carries the LDS byte address.
static __device__ __forceinline__ void async_g2lds_b128(const void* gaddr,
                                                        unsigned lds_byte_addr) {
  asm volatile("global_load_async_to_lds_b128 %0, %1, off"
               :: "v"(lds_byte_addr), "v"(gaddr) : "memory");
}
static __device__ __forceinline__ void wait_asynccnt0() {
  asm volatile("s_wait_asynccnt 0x0" ::: "memory");
}
static __device__ __forceinline__ unsigned lds_addr_of(const void* p) {
  return (unsigned)(uintptr_t)p;   // generic shared ptr: low 32 bits = LDS addr
}

// ---------------------------------------------------------------- prep kernels
__global__ __launch_bounds__(256)
void convert_x_kernel(const float* __restrict__ x, __bf16* __restrict__ xb) {
  const size_t idx = (size_t)blockIdx.x * 256 + threadIdx.x;
  xb[idx] = f2bf(x[idx]);
}

__global__ __launch_bounds__(256)
void prep_weights_kernel(const float* __restrict__ Wq, const float* __restrict__ Wk,
                         const float* __restrict__ Wv, const float* __restrict__ Wo,
                         __bf16* __restrict__ WtQKV, __bf16* __restrict__ WtO) {
  const int idx  = blockIdx.x * 256 + threadIdx.x;   // 2048*512 total
  const int nrow = idx >> 9;
  const int k    = idx & 511;
  if (nrow < 512)       WtQKV[idx] = f2bf(Wq[k * 512 + nrow]);
  else if (nrow < 1024) WtQKV[idx] = f2bf(Wk[k * 512 + (nrow - 512)]);
  else if (nrow < 1536) WtQKV[idx] = f2bf(Wv[k * 512 + (nrow - 1024)]);
  else                  WtO[(nrow - 1536) * 512 + k] = f2bf(Wo[k * 512 + (nrow - 1536)]);
}

__global__ __launch_bounds__(256)
void prep_vectors_kernel(const float* bq, const float* bk, const float* bv,
                         const float* gq, const float* gk, const float* gv,
                         const float* betaq, const float* betak, const float* betav,
                         float* bQ, float* gQ, float* betaQ,
                         float* sumQ, float* ssqQ, float* sumO, float* ssqO) {
  const int i = blockIdx.x * 256 + threadIdx.x;
  if (i < DQKV) {
    const int grp = i >> 9, c = i & 511;
    const float* bp = (grp == 0) ? bq    : (grp == 1) ? bk    : bv;
    const float* gp = (grp == 0) ? gq    : (grp == 1) ? gk    : gv;
    const float* tp = (grp == 0) ? betaq : (grp == 1) ? betak : betav;
    bQ[i] = bp[c]; gQ[i] = gp[c]; betaQ[i] = tp[c];
    sumQ[i] = 0.f; ssqQ[i] = 0.f;
  }
  if (i < D_SZ) { sumO[i] = 0.f; ssqO[i] = 0.f; }
}

// ------------------------------------------------ bf16 WMMA GEMM + BN statistics
// C = A(M x 512, bf16 row-major) * Bt^T (Bt is Ncols x 512, bf16 n-major) + bias
// Block: 256 thr (8 waves), tile 128(M) x 64(N), waves 4x2, each wave 32x32.
// Staging uses GLOBAL_LOAD_ASYNC_TO_LDS_B128 (ASYNCcnt) instead of
// load->wait->ds_store round-trips.
__global__ __launch_bounds__(256)
void gemm_bias_stats_kernel(const __bf16* __restrict__ A, const __bf16* __restrict__ Bt,
                            const float* __restrict__ bias, float* __restrict__ Y,
                            float* __restrict__ bnSum, float* __restrict__ bnSsq,
                            int Ncols) {
  __shared__ __bf16 sA[128 * 40];   // 32 k + 8 pad
  __shared__ __bf16 sB[64 * 40];
  const int tid  = threadIdx.x;
  const int lane = tid & 31;
  const int wave = tid >> 5;
  const int waveM = wave & 3;
  const int waveN = wave >> 2;
  const int lo = lane & 15;
  const int hi = (lane < 16) ? 0 : 1;
  const long m0 = (long)blockIdx.x * 128;
  const long n0 = (long)blockIdx.y * 64;

  // Per-thread staging assignments (fixed across K loop).
  const int rA0 = tid >> 2,         segA0 = tid & 3;          // A units 0..255
  const int rA1 = (tid + 256) >> 2, segA1 = tid & 3;          // A units 256..511
  const int rB  = tid >> 2,         segB  = tid & 3;          // B units 0..255
  const unsigned ldsA0 = lds_addr_of(&sA[rA0 * 40 + segA0 * 8]);
  const unsigned ldsA1 = lds_addr_of(&sA[rA1 * 40 + segA1 * 8]);
  const unsigned ldsB  = lds_addr_of(&sB[rB  * 40 + segB  * 8]);

  v8f acc[2][2] = {};
  for (int kb = 0; kb < 512; kb += 32) {
    async_g2lds_b128(A  + (m0 + rA0) * 512 + kb + segA0 * 8, ldsA0);
    async_g2lds_b128(A  + (m0 + rA1) * 512 + kb + segA1 * 8, ldsA1);
    async_g2lds_b128(Bt + (n0 + rB)  * 512 + kb + segB  * 8, ldsB);
    if (kb + 32 < 512)                             // global_prefetch_b8 next slab
      __builtin_prefetch(A + (m0 + (tid >> 1)) * 512 + kb + 32, 0, 0);
    wait_asynccnt0();
    __syncthreads();

    v16bf af[2], bfm[2];
    #pragma unroll
    for (int mt = 0; mt < 2; ++mt) {               // A frag: K chunks {0,16}+hi*8
      const __bf16* rp = &sA[(waveM * 32 + mt * 16 + lo) * 40 + hi * 8];
      af[mt] = cat8(*(const v8bf*)rp, *(const v8bf*)(rp + 16));
    }
    #pragma unroll
    for (int nt = 0; nt < 2; ++nt) {               // B frag: K chunks {0,16}+hi*8
      const __bf16* rp = &sB[(waveN * 32 + nt * 16 + lo) * 40 + hi * 8];
      bfm[nt] = cat8(*(const v8bf*)rp, *(const v8bf*)(rp + 16));
    }
    #pragma unroll
    for (int mt = 0; mt < 2; ++mt)
      #pragma unroll
      for (int nt = 0; nt < 2; ++nt)
        acc[mt][nt] = __builtin_amdgcn_wmma_f32_16x16x32_bf16(
            false, af[mt], false, bfm[nt], (short)0, acc[mt][nt], false, false);
    __syncthreads();
  }

  // Epilogue: bias, store fp32 Y, per-channel sum / sum^2 via atomics.
  #pragma unroll
  for (int nt = 0; nt < 2; ++nt) {
    const long n = n0 + waveN * 32 + nt * 16 + lo;
    const float bv = bias[n];
    float s = 0.f, s2 = 0.f;
    #pragma unroll
    for (int mt = 0; mt < 2; ++mt) {
      const long rbase = m0 + waveM * 32 + mt * 16 + hi * 8;
      #pragma unroll
      for (int p = 0; p < 8; ++p) {
        const float v = acc[mt][nt][p] + bv;
        Y[(size_t)(rbase + p) * (size_t)Ncols + n] = v;
        s += v; s2 += v * v;
      }
    }
    s  += __shfl_xor(s, 16);
    s2 += __shfl_xor(s2, 16);
    if (lane < 16) {
      atomicAdd(&bnSum[n], s);
      atomicAdd(&bnSsq[n], s2);
    }
  }
}

__global__ __launch_bounds__(256)
void bn_finalize_kernel(const float* __restrict__ s, const float* __restrict__ s2,
                        const float* __restrict__ g, const float* __restrict__ beta,
                        float* __restrict__ scale, float* __restrict__ shift, int C) {
  const int c = blockIdx.x * 256 + threadIdx.x;
  if (c < C) {
    const float invM = 1.0f / (float)M_ROWS;
    const float mean = s[c] * invM;
    const float var  = fmaxf(s2[c] * invM - mean * mean, 0.f);
    const float sc   = g[c] * rsqrtf(var + 1e-5f);
    scale[c] = sc;
    shift[c] = beta[c] - mean * sc;
  }
}

// Normalize + ReLU + split heads into WMMA-friendly layouts:
//  qg,kg : (bhn, Tpad=32, d=64) row-major bf16 (zero padded t>=24)
//  vtg   : (bhn, d=64, Tpad=32) = V^T, bf16 (zero padded)
__global__ __launch_bounds__(256)
void reshape_qkv_kernel(const float* __restrict__ Y, const float* __restrict__ scale,
                        const float* __restrict__ shift, __bf16* __restrict__ qg,
                        __bf16* __restrict__ kg, __bf16* __restrict__ vtg) {
  const size_t idx = (size_t)blockIdx.x * 256 + threadIdx.x;  // BHN*32*64
  const int dd  = (int)(idx & 63);
  const int t   = (int)((idx >> 6) & 31);
  const int bhn = (int)(idx >> 11);
  const int n   = bhn % N_SZ;
  const int rem = bhn / N_SZ;
  const int head = rem & 7;
  const int b    = rem >> 3;
  __bf16 qv = f2bf(0.f), kv = qv, vv = qv;
  if (t < T_SZ) {
    const size_t rowb = (size_t)((b * T_SZ + t) * N_SZ + n) * DQKV;
    const int cq = head * 64 + dd;
    const float q = Y[rowb + cq]         * scale[cq]         + shift[cq];
    const float k = Y[rowb + 512 + cq]   * scale[512 + cq]   + shift[512 + cq];
    const float v = Y[rowb + 1024 + cq]  * scale[1024 + cq]  + shift[1024 + cq];
    qv = f2bf(fmaxf(q, 0.f));
    kv = f2bf(fmaxf(k, 0.f));
    vv = f2bf(fmaxf(v, 0.f));
  }
  const size_t base = (size_t)bhn * (TPAD * D_HEAD);
  qg[base + t * 64 + dd]  = qv;
  kg[base + t * 64 + dd]  = kv;
  vtg[base + dd * 32 + t] = vv;
}

// ---------------------------------------------------------- per-wave attention
// One wave per (b, head, n): S = QK^T/8 (2x2 WMMA tiles, K=64), causal mask,
// softmax via 16-lane shuffles, attn->LDS->A-frags, O = attn * V (2x4 WMMA).
__global__ __launch_bounds__(256)
void attention_kernel(const __bf16* __restrict__ qg, const __bf16* __restrict__ kg,
                      const __bf16* __restrict__ vtg, __bf16* __restrict__ att) {
  __shared__ __bf16 sattn[8][TPAD * TPAD];
  const int tid = threadIdx.x, lane = tid & 31, wave = tid >> 5;
  const int lo = lane & 15, hi = (lane < 16) ? 0 : 1;
  const int bhn  = blockIdx.x * 8 + wave;
  const int n    = bhn % N_SZ;
  const int rem  = bhn / N_SZ;
  const int head = rem & 7;
  const int b    = rem >> 3;

  const __bf16* qb  = qg  + (size_t)bhn * (TPAD * D_HEAD);
  const __bf16* kbp = kg  + (size_t)bhn * (TPAD * D_HEAD);
  const __bf16* vb  = vtg + (size_t)bhn * (D_HEAD * TPAD);

  v16bf qa[2][2], kf[2][2];
  #pragma unroll
  for (int mt = 0; mt < 2; ++mt)
    #pragma unroll
    for (int ks = 0; ks < 2; ++ks) {
      const __bf16* rp = qb + (mt * 16 + lo) * 64 + ks * 32 + hi * 8;
      qa[mt][ks] = cat8(*(const v8bf*)rp, *(const v8bf*)(rp + 16));
    }
  #pragma unroll
  for (int nt = 0; nt < 2; ++nt)
    #pragma unroll
    for (int ks = 0; ks < 2; ++ks) {
      const __bf16* rp = kbp + (nt * 16 + lo) * 64 + ks * 32 + hi * 8;
      kf[nt][ks] = cat8(*(const v8bf*)rp, *(const v8bf*)(rp + 16));
    }

  v8f sc[2][2] = {};
  #pragma unroll
  for (int mt = 0; mt < 2; ++mt)
    #pragma unroll
    for (int nt = 0; nt < 2; ++nt) {
      sc[mt][nt] = __builtin_amdgcn_wmma_f32_16x16x32_bf16(
          false, qa[mt][0], false, kf[nt][0], (short)0, sc[mt][nt], false, false);
      sc[mt][nt] = __builtin_amdgcn_wmma_f32_16x16x32_bf16(
          false, qa[mt][1], false, kf[nt][1], (short)0, sc[mt][nt], false, false);
    }

  const float MASKF = -32767.0f;                  // -2^15 + 1
  #pragma unroll
  for (int mt = 0; mt < 2; ++mt) {
    #pragma unroll
    for (int p = 0; p < 8; ++p) {
      const int tg = mt * 16 + p + hi * 8;        // query row of this element
      float v0 = sc[mt][0][p] * 0.125f;           // 1/sqrt(64)
      float v1 = sc[mt][1][p] * 0.125f;
      if (lo > tg)      v0 = MASKF;               // key col = lo
      if (16 + lo > tg) v1 = MASKF;               // key col = 16+lo
      float m = fmaxf(v0, v1);
      #pragma unroll
      for (int o = 1; o < 16; o <<= 1) m = fmaxf(m, __shfl_xor(m, o));
      const float e0 = __expf(v0 - m), e1 = __expf(v1 - m);
      float s = e0 + e1;
      #pragma unroll
      for (int o = 1; o < 16; o <<= 1) s += __shfl_xor(s, o);
      const float inv = 1.0f / s;
      sattn[wave][tg * TPAD + lo]      = f2bf(e0 * inv);
      sattn[wave][tg * TPAD + 16 + lo] = f2bf(e1 * inv);
    }
  }
  __syncthreads();

  v16bf aa[2], vf[4];
  #pragma unroll
  for (int mt = 0; mt < 2; ++mt) {
    const __bf16* rp = &sattn[wave][(mt * 16 + lo) * TPAD + hi * 8];
    aa[mt] = cat8(*(const v8bf*)rp, *(const v8bf*)(rp + 16));
  }
  #pragma unroll
  for (int nt = 0; nt < 4; ++nt) {                // V^T: s chunks {0,16}+hi*8 per col
    const __bf16* rp = vb + (nt * 16 + lo) * TPAD + hi * 8;
    vf[nt] = cat8(*(const v8bf*)rp, *(const v8bf*)(rp + 16));
  }
  v8f out[2][4] = {};
  #pragma unroll
  for (int mt = 0; mt < 2; ++mt)
    #pragma unroll
    for (int nt = 0; nt < 4; ++nt)
      out[mt][nt] = __builtin_amdgcn_wmma_f32_16x16x32_bf16(
          false, aa[mt], false, vf[nt], (short)0, out[mt][nt], false, false);

  #pragma unroll
  for (int mt = 0; mt < 2; ++mt)
    #pragma unroll
    for (int p = 0; p < 8; ++p) {
      const int t = mt * 16 + p + hi * 8;
      if (t < T_SZ) {
        const size_t row = (size_t)(b * T_SZ + t) * N_SZ + n;
        #pragma unroll
        for (int nt = 0; nt < 4; ++nt)
          att[row * D_SZ + head * D_HEAD + nt * 16 + lo] = f2bf(out[mt][nt][p]);
      }
    }
}

__global__ __launch_bounds__(256)
void finalize_out_kernel(const float* __restrict__ Yo, const float* __restrict__ scale,
                         const float* __restrict__ shift, float* __restrict__ out) {
  const size_t idx = (size_t)blockIdx.x * 256 + threadIdx.x;
  const int c = (int)(idx & 511);
  out[idx] = fmaxf(Yo[idx] * scale[c] + shift[c], 0.f);
}

// ------------------------------------------------------------------- launcher
extern "C" void kernel_launch(void* const* d_in, const int* in_sizes, int n_in,
                              void* d_out, int out_size, void* d_ws, size_t ws_size,
                              hipStream_t stream) {
  (void)in_sizes; (void)n_in; (void)out_size; (void)ws_size;
  const float* x     = (const float*)d_in[0];
  const float* Wq    = (const float*)d_in[1];
  const float* bq    = (const float*)d_in[2];
  const float* gq    = (const float*)d_in[3];
  const float* betaq = (const float*)d_in[4];
  const float* Wk    = (const float*)d_in[5];
  const float* bk    = (const float*)d_in[6];
  const float* gk    = (const float*)d_in[7];
  const float* betak = (const float*)d_in[8];
  const float* Wv    = (const float*)d_in[9];
  const float* bv    = (const float*)d_in[10];
  const float* gv    = (const float*)d_in[11];
  const float* betav = (const float*)d_in[12];
  const float* Wo    = (const float*)d_in[13];
  const float* bo    = (const float*)d_in[14];
  const float* go    = (const float*)d_in[15];
  const float* betao = (const float*)d_in[16];

  char* ws = (char*)d_ws;
  size_t off = 0;
  auto alloc = [&](size_t bytes) -> char* {
    char* p = ws + off;
    off += (bytes + 255) & ~(size_t)255;
    return p;
  };
  __bf16* xb     = (__bf16*)alloc((size_t)M_ROWS * D_SZ * 2);
  __bf16* WtQKV  = (__bf16*)alloc((size_t)DQKV * D_SZ * 2);
  __bf16* WtO    = (__bf16*)alloc((size_t)D_SZ * D_SZ * 2);
  float*  bQ     = (float*)alloc(DQKV * 4);
  float*  gQ     = (float*)alloc(DQKV * 4);
  float*  betaQ  = (float*)alloc(DQKV * 4);
  float*  sumQ   = (float*)alloc(DQKV * 4);
  float*  ssqQ   = (float*)alloc(DQKV * 4);
  float*  scQ    = (float*)alloc(DQKV * 4);
  float*  shQ    = (float*)alloc(DQKV * 4);
  float*  sumO   = (float*)alloc(D_SZ * 4);
  float*  ssqO   = (float*)alloc(D_SZ * 4);
  float*  scO    = (float*)alloc(D_SZ * 4);
  float*  shO    = (float*)alloc(D_SZ * 4);
  float*  Yqkv   = (float*)alloc((size_t)M_ROWS * DQKV * 4);
  __bf16* qg     = (__bf16*)alloc((size_t)BHN * TPAD * D_HEAD * 2);
  __bf16* kg     = (__bf16*)alloc((size_t)BHN * TPAD * D_HEAD * 2);
  __bf16* vtg    = (__bf16*)alloc((size_t)BHN * D_HEAD * TPAD * 2);
  __bf16* attb   = (__bf16*)alloc((size_t)M_ROWS * D_SZ * 2);
  float*  Yo     = (float*)alloc((size_t)M_ROWS * D_SZ * 4);

  const int elemsX = M_ROWS * D_SZ;                       // 63,897,600
  convert_x_kernel<<<elemsX / 256, 256, 0, stream>>>(x, xb);
  prep_weights_kernel<<<(2048 * 512) / 256, 256, 0, stream>>>(Wq, Wk, Wv, Wo, WtQKV, WtO);
  prep_vectors_kernel<<<(DQKV + 255) / 256, 256, 0, stream>>>(
      bq, bk, bv, gq, gk, gv, betaq, betak, betav,
      bQ, gQ, betaQ, sumQ, ssqQ, sumO, ssqO);

  gemm_bias_stats_kernel<<<dim3(M_ROWS / 128, DQKV / 64), 256, 0, stream>>>(
      xb, WtQKV, bQ, Yqkv, sumQ, ssqQ, DQKV);
  bn_finalize_kernel<<<(DQKV + 255) / 256, 256, 0, stream>>>(
      sumQ, ssqQ, gQ, betaQ, scQ, shQ, DQKV);
  reshape_qkv_kernel<<<((size_t)BHN * TPAD * D_HEAD) / 256, 256, 0, stream>>>(
      Yqkv, scQ, shQ, qg, kg, vtg);

  attention_kernel<<<BHN / 8, 256, 0, stream>>>(qg, kg, vtg, attb);

  gemm_bias_stats_kernel<<<dim3(M_ROWS / 128, D_SZ / 64), 256, 0, stream>>>(
      attb, WtO, bo, Yo, sumO, ssqO, D_SZ);
  bn_finalize_kernel<<<(D_SZ + 255) / 256, 256, 0, stream>>>(
      sumO, ssqO, go, betao, scO, shO, D_SZ);
  finalize_out_kernel<<<elemsX / 256, 256, 0, stream>>>(Yo, scO, shO, (float*)d_out);
}
